// Embeddings_2D_Channel_55740085567871
// MI455X (gfx1250) — compile-verified, matrix operand order checked by reading, not run on hardware
//
#include <hip/hip_runtime.h>

// ---------------------------------------------------------------------------
// ViT patch embedding as a bf16-WMMA GEMM for gfx1250 (MI455X, wave32).
//   M = 24832 tokens, K = 256 (16x16 patch), N = 768 (E), fp32 accumulate.
//   Block: 256 threads (8 waves), tile 128(M) x 64(N), K staged fully in LDS.
// ---------------------------------------------------------------------------

typedef __attribute__((ext_vector_type(16))) __bf16        v16bf;
typedef __attribute__((ext_vector_type(8)))  float         v8f;
typedef __attribute__((ext_vector_type(4)))  unsigned int  v4u;
typedef __attribute__((ext_vector_type(2)))  unsigned int  v2u;

#define BM    128        // M rows per block
#define BN    64         // N cols per block
#define KDIM  256        // full K (one 16x16 patch)
#define LDK   264        // padded LDS row pitch in bf16 elems (16B pad: kills full conflicts)

// round-to-nearest-even fp32 -> bf16, packed as two 16-bit lanes in one dword
static __device__ __forceinline__ unsigned int pack_bf16x2(float lo, float hi) {
  unsigned int ul = __float_as_uint(lo);
  unsigned int uh = __float_as_uint(hi);
  ul = (ul + 0x7FFFu + ((ul >> 16) & 1u)) >> 16;
  uh = (uh + 0x7FFFu + ((uh >> 16) & 1u)) & 0xFFFF0000u;
  return ul | uh;
}

__global__ __launch_bounds__(256)
void patch_embed_wmma(const float* __restrict__ x,    // [64,97,32,32,1]
                      const float* __restrict__ Wm,   // [768,256]
                      const float* __restrict__ bias, // [768]
                      const float* __restrict__ pos,  // [389,768]
                      float* __restrict__ out)        // [64,389,768]
{
  extern __shared__ unsigned short smem[];
  unsigned short* lA = smem;               // BM x LDK bf16
  unsigned short* lB = smem + BM * LDK;    // BN x LDK bf16

  const int tid = threadIdx.x;
  const int n0  = blockIdx.x * BN;         // 12 blocks  -> N = 768
  const int m0  = blockIdx.y * BM;         // 194 blocks -> M = 24832

  // ---- Stage A tile: 128 tokens x 256 K, fp32 -> bf16, row-major over K ----
  // token r = b*388 + c*4 + (ph*2 + pw); K = i*16 + j  ->  x[b,c,ph*16+i,pw*16+j]
  for (int it = 0; it < 32; ++it) {
    int e4  = it * 256 + tid;              // float4 index; 64 float4 per row
    int row = e4 >> 6;
    int k   = (e4 & 63) << 2;
    int r   = m0 + row;
    int bb  = r / 388;
    int rem = r - bb * 388;
    int c   = rem >> 2;
    int p   = rem & 3;
    const float* g = x + ((((bb * 97 + c) * 32) + (p >> 1) * 16 + (k >> 4)) * 32
                          + (p & 1) * 16 + (k & 15));
    float4 v = *(const float4*)g;          // 16B load, never straddles a patch row
    v2u pk;
    pk.x = pack_bf16x2(v.x, v.y);
    pk.y = pack_bf16x2(v.z, v.w);
    *(v2u*)(lA + row * LDK + k) = pk;      // ds_store_b64
  }

  // ---- Stage B tile: B[k][n] = W[n][k]; store per-column K-contiguous ----
  for (int it = 0; it < 16; ++it) {
    int e4 = it * 256 + tid;
    int nl = e4 >> 6;
    int k  = (e4 & 63) << 2;
    const float* g = Wm + (size_t)(n0 + nl) * KDIM + k;
    float4 v = *(const float4*)g;
    v2u pk;
    pk.x = pack_bf16x2(v.x, v.y);
    pk.y = pack_bf16x2(v.z, v.w);
    *(v2u*)(lB + nl * LDK + k) = pk;
  }

  __syncthreads();

  // ---- Compute: wave w -> rows [w*16, w*16+16), 4 n-subtiles of 16 ----
  const int wave = tid >> 5;
  const int lane = tid & 31;
  const int lh   = lane >> 4;              // half-wave selects K%16 in [0,8) / [8,16)
  const int li   = lane & 15;

  v8f acc[4] = {};

  const unsigned short* pa = lA + (wave * 16 + li) * LDK + lh * 8;
  const unsigned short* pb = lB + li * LDK + lh * 8;

#pragma unroll
  for (int kb = 0; kb < 8; ++kb) {         // K = 8 x 32
    union { v16bf v; v4u q[2]; } fa;
    fa.q[0] = *(const v4u*)(pa + kb * 32);        // VGPR0-3: K = kb*32 + lh*8 + 0..7
    fa.q[1] = *(const v4u*)(pa + kb * 32 + 16);   // VGPR4-7: +16 (second patch row)
#pragma unroll
    for (int t = 0; t < 4; ++t) {
      union { v16bf v; v4u q[2]; } fb;
      const unsigned short* pbt = pb + t * 16 * LDK + kb * 32;
      fb.q[0] = *(const v4u*)(pbt);
      fb.q[1] = *(const v4u*)(pbt + 16);
      acc[t] = __builtin_amdgcn_wmma_f32_16x16x32_bf16(
          false, fa.v, false, fb.v, (short)0, acc[t], false, false);
    }
  }

  // ---- Epilogue: + bias + pos, scatter to [b, 1+tok, e] ----
#pragma unroll
  for (int t = 0; t < 4; ++t) {
    int e = n0 + t * 16 + li;              // lanes 0-15: N=e ; lanes 16-31: same N, M+8
    float bv = bias[e];
#pragma unroll
    for (int v = 0; v < 8; ++v) {          // C/D layout: VGPR v -> M = v + lh*8
      int r   = m0 + wave * 16 + lh * 8 + v;
      int bb  = r / 388;
      int tok = r - bb * 388 + 1;          // +1: cls token occupies slot 0
      out[((size_t)bb * 389 + tok) * 768 + e] = acc[t][v] + bv + pos[(size_t)tok * 768 + e];
    }
  }
}

// cls row: out[b, 0, e] = cls[e] + pos[0, e]
__global__ __launch_bounds__(256)
void cls_pos_kernel(const float* __restrict__ cls, const float* __restrict__ pos,
                    float* __restrict__ out) {
  int i = blockIdx.x * 256 + threadIdx.x;
  if (i >= 64 * 768) return;
  int bb = i / 768;
  int e  = i - bb * 768;
  out[(size_t)bb * 389 * 768 + e] = cls[e] + pos[e];
}

extern "C" void kernel_launch(void* const* d_in, const int* in_sizes, int n_in,
                              void* d_out, int out_size, void* d_ws, size_t ws_size,
                              hipStream_t stream) {
  const float* x    = (const float*)d_in[0];  // [64,97,32,32,1]
  const float* Wm   = (const float*)d_in[1];  // [768,1,16,16]
  const float* bias = (const float*)d_in[2];  // [768]
  const float* pos  = (const float*)d_in[3];  // [1,389,768]
  const float* cls  = (const float*)d_in[4];  // [1,1,768]
  float* out        = (float*)d_out;          // [64,389,768]

  const size_t shmem = (size_t)(BM * LDK + BN * LDK) * sizeof(unsigned short); // ~99 KB
  patch_embed_wmma<<<dim3(12, 194), 256, shmem, stream>>>(x, Wm, bias, pos, out);
  cls_pos_kernel<<<dim3(192), 256, 0, stream>>>(cls, pos, out);
}
